// Transformer_Layer_51067161149566
// MI455X (gfx1250) — matrix operand order repaired and checked by reading, not compiled, not run on hardware
//
#include <hip/hip_runtime.h>

// ---------------------------------------------------------------------------
// Problem constants (reference: D=768, H=3, x is (4, 2048, 768) fp32)
// ---------------------------------------------------------------------------
#define DD    768
#define HH    3
#define BB    4
#define SS    2048
#define MROWS (BB * SS)   // 8192 token rows
#define NQKV  (HH * DD)   // 2304

typedef __attribute__((ext_vector_type(16))) __bf16 bfrag16;  // WMMA A/B fragment
typedef __attribute__((ext_vector_type(8)))  __bf16 bvec8;
typedef __attribute__((ext_vector_type(8)))  float  facc8;    // WMMA C/D fragment
typedef __attribute__((ext_vector_type(4)))  unsigned int su32x4;  // D# group 0
typedef __attribute__((ext_vector_type(8)))  unsigned int su32x8;  // D# group 1

// A-matrix fragment (16x32 bf16), in this code's contiguous-K order (not the ISA table order):
//   lane group g = lane>>4, row M = lane&15
//   elements 0..15 = K = g*16 + 0..15
// caller passes p = base + row*strideK + kchunk + g*16   (32B aligned)
__device__ __forceinline__ bfrag16 load_fragA(const __bf16* p) {
    return *(const bfrag16*)p;
}

// B-matrix fragment (32x16 bf16): lanes 0..15 hold K=0..15, lanes 16..31 hold
// K=16..31, col N = lane&15.  p = Bt + n*strideK + kchunk + g*16  (32B aligned)
__device__ __forceinline__ bfrag16 load_fragB(const __bf16* p) {
    return *(const bfrag16*)p;
}

__device__ __forceinline__ facc8 wmma_bf16(bfrag16 a, bfrag16 b, facc8 c) {
    return __builtin_amdgcn_wmma_f32_16x16x32_bf16(false, a, false, b,
                                                   (short)0, c, false, false);
}

// ---------------------------------------------------------------------------
// Tensor Data Mover: async-DMA a 2D bf16 tile (tile_w x tile_h, row stride
// stride_elems) from global into LDS at lds_off.  D# built per ISA 08 §8.3/8.4:
//   group0: [1:0]=count=1, [63:32]=lds_addr, [120:64]=global_addr, [127:126]=2
//   group1: [17:16]=data_size(1=2B), tensor_dim0/1, tile_dim0/1, dim0_stride
// Issued per-wave (EXEC ignored); completion tracked by TENSORcnt.
// ---------------------------------------------------------------------------
__device__ __forceinline__ void tdm_load_2d_bf16(unsigned lds_off,
                                                 const void* gptr,
                                                 unsigned tile_w, unsigned tile_h,
                                                 unsigned stride_elems,
                                                 unsigned tensor_w, unsigned tensor_h) {
    unsigned long long ga = (unsigned long long)(uintptr_t)gptr;
    su32x4 g0;
    g0[0] = 1u;                                     // count=1, user descriptor
    g0[1] = lds_off;                                // lds_addr (bytes)
    g0[2] = (unsigned)ga;                           // global_addr[31:0]
    g0[3] = (unsigned)((ga >> 32) & 0x1FFFFFFu)     // global_addr[56:32]
          | (2u << 30);                             // type = 2 ("image")
    su32x8 g1;
    g1[0] = 1u << 16;                               // wg_mask=0, data_size=2B
    g1[1] = (tensor_w & 0xFFFFu) << 16;             // tensor_dim0[15:0] @ bit48
    g1[2] = (tensor_w >> 16) | ((tensor_h & 0xFFFFu) << 16);   // dim0[31:16], dim1[15:0]
    g1[3] = (tensor_h >> 16) | (tile_w << 16);      // dim1[31:16], tile_dim0
    g1[4] = tile_h & 0xFFFFu;                       // tile_dim1 (tile_dim2=0)
    g1[5] = stride_elems;                           // tensor_dim0_stride[31:0]
    g1[6] = 0u;                                     // stride0[47:32], stride1 lo
    g1[7] = 0u;                                     // stride1 hi
    asm volatile("tensor_load_to_lds %0, %1" :: "s"(g0), "s"(g1) : "memory");
}

// ---------------------------------------------------------------------------
// Elementwise helpers
// ---------------------------------------------------------------------------
__global__ void cvt_f32_to_bf16(const float* __restrict__ in,
                                __bf16* __restrict__ out, int n) {
    for (int i = blockIdx.x * blockDim.x + threadIdx.x; i < n;
         i += gridDim.x * blockDim.x)
        out[i] = (__bf16)in[i];
}

// in: fp32 [R][C]  ->  out: bf16 [C][R]   (B-transpose for contiguous-K frags)
__global__ void transpose_cvt(const float* __restrict__ in,
                              __bf16* __restrict__ out, int R, int C) {
    int n = R * C;
    for (int i = blockIdx.x * blockDim.x + threadIdx.x; i < n;
         i += gridDim.x * blockDim.x) {
        int r = i / C, c = i - r * C;
        out[(size_t)c * R + r] = (__bf16)in[i];
    }
}

__global__ void reduce_sum_f32(const float* __restrict__ a,
                               float* __restrict__ out, int n) {
    float s = 0.f;
    for (int i = blockIdx.x * blockDim.x + threadIdx.x; i < n;
         i += gridDim.x * blockDim.x) s += a[i];
    for (int m = 16; m >= 1; m >>= 1) s += __shfl_xor(s, m);
    __shared__ float ls[8];
    int lane = threadIdx.x & 31, wid = threadIdx.x >> 5;
    if (lane == 0) ls[wid] = s;
    __syncthreads();
    if (wid == 0) {
        s = (lane < (int)(blockDim.x >> 5)) ? ls[lane] : 0.f;
        for (int m = 4; m >= 1; m >>= 1) s += __shfl_xor(s, m);
        if (lane == 0) atomicAdd(out, s);
    }
}

// yb = bf16( a - mean(a) + x )
__global__ void residual_meansub(const float* __restrict__ a,
                                 const float* __restrict__ x,
                                 const float* __restrict__ gsum,
                                 __bf16* __restrict__ yb, int n, float invN) {
    float mean = (*gsum) * invN;
    for (int i = blockIdx.x * blockDim.x + threadIdx.x; i < n;
         i += gridDim.x * blockDim.x)
        yb[i] = (__bf16)(a[i] - mean + x[i]);
}

// ---------------------------------------------------------------------------
// WMMA GEMM with TDM-staged, double-buffered LDS tiles:
//   C[M,N] = A[M,K] @ Bt[N,K]^T + bias[N]
//   block = 256 threads (8 waves), tile 64(M) x 256(N); wave tile 32x64
//   Per 32-wide K step, wave 0 DMAs A(64x32) and B(256x32) tiles to LDS via
//   tensor_load_to_lds; all waves consume shared tiles (A reused 4x, B 2x).
//   MODE 0: fp32 out   MODE 1: bf16 out
//   MODE 2: bf16 out, V transposed per head -> Vt[b][h][d][s]
//   MODE 3: fp32 out with leaky_relu(0.01) epilogue
// ---------------------------------------------------------------------------
template <int MODE>
__global__ __launch_bounds__(256)
void gemm_bf16_wmma(const __bf16* __restrict__ A, const __bf16* __restrict__ Bt,
                    const float* __restrict__ bias, void* __restrict__ Cout,
                    int M, int N, int K) {
    __shared__ __align__(64) __bf16 ldsA[2][64 * 32];    //  2 x 4 KB
    __shared__ __align__(64) __bf16 ldsB[2][256 * 32];   //  2 x 16 KB

    const int lane = threadIdx.x & 31;
    const int wid  = threadIdx.x >> 5;
    const int g    = lane >> 4;
    const int lm   = lane & 15;
    const int wm   = wid >> 2;      // 0..1
    const int wn   = wid & 3;       // 0..3
    const int m0   = blockIdx.y * 64;
    const int n0   = blockIdx.x * 256;

    facc8 acc[2][4] = {};

    const int nk = K / 32;
    if (wid == 0) {  // prologue: stage k-chunk 0 into buffer 0
        tdm_load_2d_bf16((unsigned)(uintptr_t)&ldsA[0][0], A + (size_t)m0 * K,
                         32, 64, (unsigned)K, (unsigned)K, (unsigned)(M - m0));
        tdm_load_2d_bf16((unsigned)(uintptr_t)&ldsB[0][0], Bt + (size_t)n0 * K,
                         32, 256, (unsigned)K, (unsigned)K, (unsigned)(N - n0));
    }

    for (int ks = 0; ks < nk; ++ks) {
        const int cur = ks & 1;
        if (wid == 0) __builtin_amdgcn_s_wait_tensorcnt(0);
        __syncthreads();                     // tiles[cur] visible to all waves
        if (wid == 0 && ks + 1 < nk) {       // prefetch next chunk into buf cur^1
            const int kc = (ks + 1) * 32;    // (buf cur^1 was consumed pre-barrier)
            tdm_load_2d_bf16((unsigned)(uintptr_t)&ldsA[cur ^ 1][0],
                             A + (size_t)m0 * K + kc, 32, 64,
                             (unsigned)K, (unsigned)(K - kc), (unsigned)(M - m0));
            tdm_load_2d_bf16((unsigned)(uintptr_t)&ldsB[cur ^ 1][0],
                             Bt + (size_t)n0 * K + kc, 32, 256,
                             (unsigned)K, (unsigned)(K - kc), (unsigned)(N - n0));
        }

        bfrag16 af[2], bfg[4];
        for (int tm = 0; tm < 2; ++tm)
            af[tm] = load_fragA(&ldsA[cur][(wm * 32 + tm * 16 + lm) * 32 + g * 16]);
        for (int tn = 0; tn < 4; ++tn)
            bfg[tn] = load_fragB(&ldsB[cur][(wn * 64 + tn * 16 + lm) * 32 + g * 16]);
        for (int tm = 0; tm < 2; ++tm)
            for (int tn = 0; tn < 4; ++tn)
                acc[tm][tn] = wmma_bf16(af[tm], bfg[tn], acc[tm][tn]);
        __syncthreads();                     // done reading tiles[cur]
    }

    const int mo = g * 8;  // C layout: VGPR r -> row r (lanes 0-15) / r+8 (16-31)
    for (int tm = 0; tm < 2; ++tm)
        for (int tn = 0; tn < 4; ++tn) {
            const int nn  = n0 + wn * 64 + tn * 16 + lm;
            const float bv = bias[nn];
            for (int r = 0; r < 8; ++r) {
                const int mm = m0 + wm * 32 + tm * 16 + mo + r;
                float v = acc[tm][tn][r] + bv;
                if (MODE == 0) {
                    ((float*)Cout)[(size_t)mm * N + nn] = v;
                } else if (MODE == 1) {
                    ((__bf16*)Cout)[(size_t)mm * N + nn] = (__bf16)v;
                } else if (MODE == 2) {
                    const int b = mm >> 11, s = mm & (SS - 1);
                    const int h = nn / DD,  d = nn - h * DD;
                    ((__bf16*)Cout)[(((size_t)b * HH + h) * DD + d) * SS + s] =
                        (__bf16)v;
                } else {  // MODE 3
                    v = v > 0.f ? v : 0.01f * v;
                    ((float*)Cout)[(size_t)mm * N + nn] = v;
                }
            }
        }
}

// ---------------------------------------------------------------------------
// Flash attention, head dim = 768, unscaled scores, faithful mask.
//   grid = (128 q-tiles, B*H); block = 128 threads (4 waves)
//   wave w owns d-slice [w*192, w*192+192) -> 12 output n-tiles
//   Direct-from-global fragments: per-(b,h) K/V set (~6 MB) is L2-resident.
// ---------------------------------------------------------------------------
__global__ __launch_bounds__(128)
void attn_flash(const __bf16* __restrict__ Q, const __bf16* __restrict__ Kb,
                const __bf16* __restrict__ Vt, __bf16* __restrict__ O) {
    __shared__ __align__(32) __bf16 Plds[4][16][32];  // per-wave P relayout patch

    const int lane = threadIdx.x & 31;
    const int w    = threadIdx.x >> 5;
    const int g    = lane >> 4;
    const int lm   = lane & 15;
    const int qt   = blockIdx.x;
    const int bh   = blockIdx.y;
    const int b    = bh / HH, h = bh - b * HH;
    const int q0   = qt * 16;

    const __bf16* qrow  = Q  + (size_t)(b * SS + q0 + lm) * NQKV + h * DD + g * 16;
    const __bf16* kbase = Kb + (size_t)(b * SS) * NQKV + h * DD + g * 16;
    const __bf16* vbase = Vt + (size_t)bh * DD * SS;

    facc8 acc[12] = {};
    float mrow[8], lrow[8];
    for (int r = 0; r < 8; ++r) { mrow[r] = -3.0e38f; lrow[r] = 0.f; }

    const int nsteps = (q0 + 16 + 31) >> 5;   // causal: keys 0 .. q0+15
    for (int ks = 0; ks < nsteps; ++ks) {
        const int k0 = ks * 32;

        facc8 s0 = {}, s1 = {};
        for (int kc = 0; kc < DD; kc += 32) {
            bfrag16 aq  = load_fragA(qrow + kc);
            bfrag16 bk0 = load_fragB(kbase + (size_t)(k0 + lm)      * NQKV + kc);
            bfrag16 bk1 = load_fragB(kbase + (size_t)(k0 + 16 + lm) * NQKV + kc);
            s0 = wmma_bf16(aq, bk0, s0);
            s1 = wmma_bf16(aq, bk1, s1);
        }

        // faithful mask: tril(scores)==0 -> -99999 (col>row OR score==0)
        const int n0k = k0 + lm, n1k = k0 + 16 + lm;
        float p0[8], p1[8];
        for (int r = 0; r < 8; ++r) {
            const int qq = q0 + g * 8 + r;
            float v0 = s0[r]; p0[r] = (n0k > qq || v0 == 0.f) ? -99999.f : v0;
            float v1 = s1[r]; p1[r] = (n1k > qq || v1 == 0.f) ? -99999.f : v1;
        }

        // online softmax
        for (int r = 0; r < 8; ++r) {
            float rmax = fmaxf(p0[r], p1[r]);
            for (int m = 8; m >= 1; m >>= 1) rmax = fmaxf(rmax, __shfl_xor(rmax, m));
            const float mnew = fmaxf(mrow[r], rmax);
            const float resc = __expf(mrow[r] - mnew);
            const float e0 = __expf(p0[r] - mnew);
            const float e1 = __expf(p1[r] - mnew);
            float rsum = e0 + e1;
            for (int m = 8; m >= 1; m >>= 1) rsum += __shfl_xor(rsum, m);
            lrow[r] = lrow[r] * resc + rsum;
            mrow[r] = mnew;
            p0[r] = e0; p1[r] = e1;
            for (int t = 0; t < 12; ++t) acc[t][r] *= resc;
        }

        // C-layout -> A-layout relayout of P through per-wave LDS patch
        for (int r = 0; r < 8; ++r) {
            const int row = g * 8 + r;
            Plds[w][row][lm]      = (__bf16)p0[r];
            Plds[w][row][lm + 16] = (__bf16)p1[r];
        }

        // O += P(16x32) @ V(32 x d-slice)
        bfrag16 ap = load_fragA(&Plds[w][lm][g * 16]);
        for (int t = 0; t < 12; ++t) {
            const int dn = w * 192 + t * 16 + lm;
            bfrag16 bv = load_fragB(vbase + (size_t)dn * SS + k0 + g * 16);
            acc[t] = wmma_bf16(ap, bv, acc[t]);
        }
    }

    for (int t = 0; t < 12; ++t) {
        const int dn = w * 192 + t * 16 + lm;
        for (int r = 0; r < 8; ++r) {
            const int qq = q0 + g * 8 + r;
            O[(size_t)(b * SS + qq) * NQKV + h * DD + dn] =
                (__bf16)(acc[t][r] / lrow[r]);
        }
    }
}

// ---------------------------------------------------------------------------
// Host side
// ---------------------------------------------------------------------------
extern "C" void kernel_launch(void* const* d_in, const int* in_sizes, int n_in,
                              void* d_out, int out_size, void* d_ws, size_t ws_size,
                              hipStream_t stream) {
    (void)in_sizes; (void)n_in; (void)out_size; (void)ws_size;

    const float* x   = (const float*)d_in[0];
    const float* Wq  = (const float*)d_in[1];
    const float* bq  = (const float*)d_in[2];
    const float* Wk  = (const float*)d_in[3];
    const float* bk  = (const float*)d_in[4];
    const float* Wv  = (const float*)d_in[5];
    const float* bv  = (const float*)d_in[6];
    const float* Wo  = (const float*)d_in[7];
    const float* bo  = (const float*)d_in[8];
    const float* Wf  = (const float*)d_in[9];
    const float* bfb = (const float*)d_in[10];
    float* out = (float*)d_out;

    char* p = (char*)d_ws;
    auto carve = [&](size_t bytes) -> char* {
        char* r = p; p += (bytes + 255) & ~(size_t)255; return r;
    };
    __bf16* xb   = (__bf16*)carve((size_t)MROWS * DD * 2);
    __bf16* Wqt  = (__bf16*)carve((size_t)NQKV * DD * 2);
    __bf16* Wkt  = (__bf16*)carve((size_t)NQKV * DD * 2);
    __bf16* Wvt  = (__bf16*)carve((size_t)NQKV * DD * 2);
    __bf16* Wot  = (__bf16*)carve((size_t)DD * NQKV * 2);
    __bf16* Wft  = (__bf16*)carve((size_t)DD * DD * 2);
    __bf16* Qb   = (__bf16*)carve((size_t)MROWS * NQKV * 2);
    __bf16* Kbf  = (__bf16*)carve((size_t)MROWS * NQKV * 2);
    __bf16* Vtb  = (__bf16*)carve((size_t)MROWS * NQKV * 2);
    __bf16* Ob   = (__bf16*)carve((size_t)MROWS * NQKV * 2);
    float*  abuf = (float*) carve((size_t)MROWS * DD * 4);
    __bf16* yb   = (__bf16*)carve((size_t)MROWS * DD * 2);
    float*  gsum = (float*) carve(256);

    const int nx = MROWS * DD;  // 6291456

    cvt_f32_to_bf16<<<(nx + 255) / 256, 256, 0, stream>>>(x, xb, nx);
    transpose_cvt<<<(DD * NQKV + 255) / 256, 256, 0, stream>>>(Wq, Wqt, DD, NQKV);
    transpose_cvt<<<(DD * NQKV + 255) / 256, 256, 0, stream>>>(Wk, Wkt, DD, NQKV);
    transpose_cvt<<<(DD * NQKV + 255) / 256, 256, 0, stream>>>(Wv, Wvt, DD, NQKV);
    transpose_cvt<<<(NQKV * DD + 255) / 256, 256, 0, stream>>>(Wo, Wot, NQKV, DD);
    transpose_cvt<<<(DD * DD + 255) / 256, 256, 0, stream>>>(Wf, Wft, DD, DD);

    dim3 gqkv(NQKV / 256, MROWS / 64);
    gemm_bf16_wmma<1><<<gqkv, 256, 0, stream>>>(xb, Wqt, bq, Qb,  MROWS, NQKV, DD);
    gemm_bf16_wmma<1><<<gqkv, 256, 0, stream>>>(xb, Wkt, bk, Kbf, MROWS, NQKV, DD);
    gemm_bf16_wmma<2><<<gqkv, 256, 0, stream>>>(xb, Wvt, bv, Vtb, MROWS, NQKV, DD);

    attn_flash<<<dim3(SS / 16, BB * HH), 128, 0, stream>>>(Qb, Kbf, Vtb, Ob);

    dim3 go(DD / 256, MROWS / 64);
    gemm_bf16_wmma<0><<<go, 256, 0, stream>>>(Ob, Wot, bo, abuf, MROWS, DD, NQKV);

    hipMemsetAsync(gsum, 0, sizeof(float), stream);
    reduce_sum_f32<<<256, 256, 0, stream>>>(abuf, gsum, nx);
    residual_meansub<<<(nx + 255) / 256, 256, 0, stream>>>(
        abuf, x, gsum, yb, nx, 1.0f / (float)nx);

    gemm_bf16_wmma<3><<<go, 256, 0, stream>>>(yb, Wft, bfb, out, MROWS, DD, DD);
}
